// SelfAttention_72361609003476
// MI455X (gfx1250) — compile-verified, matrix operand order checked by reading, not run on hardware
//
#include <hip/hip_runtime.h>
#include <hip/hip_bf16.h>

// ---- problem constants (match reference) ----
#define B_  4
#define S_  2048
#define D_  1024
#define H_  16
#define HD_ 64
#define M_  (B_ * S_)   // 8192

typedef __bf16 v16bf __attribute__((ext_vector_type(16)));
typedef __bf16 v8bf  __attribute__((ext_vector_type(8)));
typedef float  v8f   __attribute__((ext_vector_type(8)));

union Frag {
    v16bf v;
    v8bf  h[2];
};

__device__ __forceinline__ v8f zero8() {
    v8f z;
#pragma unroll
    for (int i = 0; i < 8; ++i) z[i] = 0.0f;
    return z;
}

// ---------------------------------------------------------------------------
// fp32 -> bf16 cast (8 elems / thread, sizes are multiples of 2048)
// ---------------------------------------------------------------------------
__global__ void __launch_bounds__(256)
attn_cast_bf16(const float* __restrict__ s, __bf16* __restrict__ d, int n) {
    int i = (blockIdx.x * blockDim.x + threadIdx.x) * 8;
    if (i >= n) return;
    float4 a = *(const float4*)(s + i);
    float4 b = *(const float4*)(s + i + 4);
    v8bf o;
    o[0] = (__bf16)a.x; o[1] = (__bf16)a.y; o[2] = (__bf16)a.z; o[3] = (__bf16)a.w;
    o[4] = (__bf16)b.x; o[5] = (__bf16)b.y; o[6] = (__bf16)b.z; o[7] = (__bf16)b.w;
    *(v8bf*)(d + i) = o;
}

// ---------------------------------------------------------------------------
// C[M,N] = A[M,K] * W[N,K]^T + bias   (bf16 WMMA, fp32 accum)
//   PERMUTE: write bf16 out permuted [B,H,S,HD]   (QKV projections)
//   F32OUT : write fp32 out row-major [M,N]       (output projection)
// Block: 256 thr = 8 waves; block tile 128x128, BK=32; wave tile 32x64.
// ---------------------------------------------------------------------------
template <bool PERMUTE, bool F32OUT>
__global__ void __launch_bounds__(256)
attn_gemm_wmma(const __bf16* __restrict__ A,
               const __bf16* __restrict__ W,
               const float* __restrict__ bias,
               void* __restrict__ out,
               int M, int N, int K) {
    constexpr int AS_LD = 40;   // 32 + 8 pad (80B rows, 16B aligned)
    __shared__ __bf16 As[128 * AS_LD];
    __shared__ __bf16 Bs[128 * AS_LD];

    const int tid  = threadIdx.x;
    const int wave = tid >> 5;
    const int lane = tid & 31;
    const int l16  = lane & 15;
    const int half = lane >> 4;
    const int kb   = half * 8;

    const int bm = blockIdx.x * 128;
    const int bn = blockIdx.y * 128;
    const int wm = (wave & 3) * 32;
    const int wn = (wave >> 2) * 64;

    v8f acc[2][4];
#pragma unroll
    for (int i = 0; i < 2; ++i)
#pragma unroll
        for (int j = 0; j < 4; ++j) acc[i][j] = zero8();

    for (int k0 = 0; k0 < K; k0 += 32) {
        // stage A and W tiles (each 128 rows x 32 bf16)
#pragma unroll
        for (int c0 = 0; c0 < 2; ++c0) {
            int c    = tid * 2 + c0;         // 0..511
            int row  = c >> 2;
            int col8 = (c & 3) * 8;
            *(v8bf*)&As[row * AS_LD + col8] =
                *(const v8bf*)&A[(size_t)(bm + row) * K + k0 + col8];
            *(v8bf*)&Bs[row * AS_LD + col8] =
                *(const v8bf*)&W[(size_t)(bn + row) * K + k0 + col8];
        }
        if (k0 + 32 < K) {
            __builtin_prefetch(&A[(size_t)(bm + tid / 2) * K + k0 + 32], 0, 0);
            __builtin_prefetch(&W[(size_t)(bn + tid / 2) * K + k0 + 32], 0, 0);
        }
        __syncthreads();

        Frag af[2], bf[4];
#pragma unroll
        for (int i = 0; i < 2; ++i) {
            int row = wm + i * 16 + l16;
            af[i].h[0] = *(v8bf*)&As[row * AS_LD + kb];
            af[i].h[1] = *(v8bf*)&As[row * AS_LD + kb + 16];
        }
#pragma unroll
        for (int j = 0; j < 4; ++j) {
            int row = wn + j * 16 + l16;
            bf[j].h[0] = *(v8bf*)&Bs[row * AS_LD + kb];
            bf[j].h[1] = *(v8bf*)&Bs[row * AS_LD + kb + 16];
        }
#pragma unroll
        for (int i = 0; i < 2; ++i)
#pragma unroll
            for (int j = 0; j < 4; ++j)
                acc[i][j] = __builtin_amdgcn_wmma_f32_16x16x32_bf16(
                    false, af[i].v, false, bf[j].v, (short)0, acc[i][j], false, false);
        __syncthreads();
    }

    // epilogue
#pragma unroll
    for (int i = 0; i < 2; ++i) {
#pragma unroll
        for (int j = 0; j < 4; ++j) {
            int n = bn + wn + j * 16 + l16;
            float bs = bias[n];
#pragma unroll
            for (int r = 0; r < 8; ++r) {
                int m = bm + wm + i * 16 + half * 8 + r;
                float val = acc[i][j][r] + bs;
                if constexpr (F32OUT) {
                    ((float*)out)[(size_t)m * N + n] = val;
                } else if constexpr (PERMUTE) {
                    int b = m / S_, s = m % S_;
                    int h = n / HD_, hd = n % HD_;
                    ((__bf16*)out)[(((size_t)b * H_ + h) * S_ + s) * HD_ + hd] = (__bf16)val;
                } else {
                    ((__bf16*)out)[(size_t)m * N + n] = (__bf16)val;
                }
            }
        }
    }
}

// ---------------------------------------------------------------------------
// Flash attention: one block = 128 queries of one (b,h). 8 waves, each
// owns 16 query rows. Online softmax over 64-key tiles (keeps register
// pressure low: 4 score frags + 4 out frags + 2 Q frags live).
// Q,K,V layout: [B,H,S,HD] bf16.  Output: [B,S,D] bf16.
// ---------------------------------------------------------------------------
__global__ void __launch_bounds__(256)
attn_flash_wmma(const __bf16* __restrict__ q,
                const __bf16* __restrict__ k,
                const __bf16* __restrict__ v,
                __bf16* __restrict__ o) {
    constexpr int QB = 128, KB = 64;
    constexpr int KS_LD = 72;    // 64 + 8 pad (144B rows, 16B aligned)
    constexpr int VT_LD = 72;    // 64 keys + 8 pad
    constexpr int PS_LD = 72;    // 64 keys + 8 pad (per-wave P staging)

    __shared__ __bf16 Ks [KB * KS_LD];        //  9 KB
    __shared__ __bf16 Vts[HD_ * VT_LD];       //  9 KB (transposed: [hd][key])
    __shared__ __bf16 Ps [8 * 16 * PS_LD];    // 18 KB

    const int tid  = threadIdx.x;
    const int wave = tid >> 5;
    const int lane = tid & 31;
    const int l16  = lane & 15;
    const int half = lane >> 4;
    const int kb   = half * 8;

    const int tilesPerBH = S_ / QB;                 // 16
    const int bh = blockIdx.x / tilesPerBH;
    const int qt = blockIdx.x % tilesPerBH;

    const __bf16* Qp = q + ((size_t)bh * S_ + qt * QB) * HD_;
    const __bf16* Kp = k + (size_t)bh * S_ * HD_;
    const __bf16* Vp = v + (size_t)bh * S_ * HD_;

    // Q fragments for this wave's 16 rows (HD=64 -> two K-chunks of 32)
    Frag qf[2];
    {
        int qrow = wave * 16 + l16;
#pragma unroll
        for (int c = 0; c < 2; ++c) {
            qf[c].h[0] = *(const v8bf*)&Qp[(size_t)qrow * HD_ + c * 32 + kb];
            qf[c].h[1] = *(const v8bf*)&Qp[(size_t)qrow * HD_ + c * 32 + kb + 16];
        }
    }

    float mrun[8], lrun[8];
    v8f oacc[4];
#pragma unroll
    for (int r = 0; r < 8; ++r) { mrun[r] = -1e30f; lrun[r] = 0.0f; }
#pragma unroll
    for (int f = 0; f < 4; ++f) oacc[f] = zero8();

    __bf16* Pw = &Ps[wave * 16 * PS_LD];

    for (int kt = 0; kt < S_ / KB; ++kt) {
        const __bf16* Kt = Kp + (size_t)kt * KB * HD_;
        const __bf16* Vt = Vp + (size_t)kt * KB * HD_;

        // stage K row-major [key][hd] and V transposed [hd][key]
        // 64 keys x 64 hd = 512 chunks of 8; 2 chunks / thread
#pragma unroll
        for (int c0 = 0; c0 < 2; ++c0) {
            int c    = tid * 2 + c0;        // 0..511
            int key  = c >> 3;
            int col8 = (c & 7) * 8;
            *(v8bf*)&Ks[key * KS_LD + col8] =
                *(const v8bf*)&Kt[(size_t)key * HD_ + col8];
            v8bf vv = *(const v8bf*)&Vt[(size_t)key * HD_ + col8];
#pragma unroll
            for (int e = 0; e < 8; ++e)
                Vts[(col8 + e) * VT_LD + key] = vv[e];
        }
        __syncthreads();

        // scores: 16 queries x 64 keys  (4 frags x 2 K-chunks = 8 WMMAs)
        v8f sacc[4];
#pragma unroll
        for (int j = 0; j < 4; ++j) sacc[j] = zero8();
#pragma unroll
        for (int j = 0; j < 4; ++j) {
            int keyc = j * 16 + l16;
#pragma unroll
            for (int c = 0; c < 2; ++c) {
                Frag bf;
                bf.h[0] = *(v8bf*)&Ks[keyc * KS_LD + c * 32 + kb];
                bf.h[1] = *(v8bf*)&Ks[keyc * KS_LD + c * 32 + kb + 16];
                sacc[j] = __builtin_amdgcn_wmma_f32_16x16x32_bf16(
                    false, qf[c].v, false, bf.v, (short)0, sacc[j], false, false);
            }
        }
#pragma unroll
        for (int j = 0; j < 4; ++j)
#pragma unroll
            for (int r = 0; r < 8; ++r) sacc[j][r] *= 0.125f;  // 1/sqrt(64)

        // row max over 64 keys (elementwise over frags, then across 16 lanes)
        float mx[8];
#pragma unroll
        for (int r = 0; r < 8; ++r) mx[r] = sacc[0][r];
#pragma unroll
        for (int j = 1; j < 4; ++j)
#pragma unroll
            for (int r = 0; r < 8; ++r) mx[r] = fmaxf(mx[r], sacc[j][r]);
#pragma unroll
        for (int mask = 1; mask <= 8; mask <<= 1)
#pragma unroll
            for (int r = 0; r < 8; ++r)
                mx[r] = fmaxf(mx[r], __shfl_xor(mx[r], mask, 32));

        // online rescale
#pragma unroll
        for (int r = 0; r < 8; ++r) {
            float mn   = fmaxf(mrun[r], mx[r]);
            float corr = __expf(mrun[r] - mn);
            mrun[r] = mn;
            lrun[r] *= corr;
#pragma unroll
            for (int f = 0; f < 4; ++f) oacc[f][r] *= corr;
        }

        // exp + stage P (bf16, per-wave region), accumulate row sums
        float sm[8];
#pragma unroll
        for (int r = 0; r < 8; ++r) sm[r] = 0.0f;
#pragma unroll
        for (int j = 0; j < 4; ++j) {
#pragma unroll
            for (int r = 0; r < 8; ++r) {
                float p = __expf(sacc[j][r] - mrun[r]);
                sm[r] += p;
                Pw[(half * 8 + r) * PS_LD + j * 16 + l16] = (__bf16)p;
            }
        }

        // P (16x64) * V (64x64): 2 K-chunks x 4 hd subtiles = 8 WMMAs
#pragma unroll
        for (int kc = 0; kc < 2; ++kc) {
            Frag pa;
            pa.h[0] = *(v8bf*)&Pw[l16 * PS_LD + kc * 32 + kb];
            pa.h[1] = *(v8bf*)&Pw[l16 * PS_LD + kc * 32 + kb + 16];
#pragma unroll
            for (int f = 0; f < 4; ++f) {
                Frag vb;
                int kcol = kc * 32 + kb;
                vb.h[0] = *(v8bf*)&Vts[(f * 16 + l16) * VT_LD + kcol];
                vb.h[1] = *(v8bf*)&Vts[(f * 16 + l16) * VT_LD + kcol + 16];
                oacc[f] = __builtin_amdgcn_wmma_f32_16x16x32_bf16(
                    false, pa.v, false, vb.v, (short)0, oacc[f], false, false);
            }
        }

#pragma unroll
        for (int mask = 1; mask <= 8; mask <<= 1)
#pragma unroll
            for (int r = 0; r < 8; ++r) sm[r] += __shfl_xor(sm[r], mask, 32);
#pragma unroll
        for (int r = 0; r < 8; ++r) lrun[r] += sm[r];
        __syncthreads();
    }

    // normalize and write O as [B,S,D] bf16
    const int b = bh / H_, h = bh % H_;
#pragma unroll
    for (int r = 0; r < 8; ++r) {
        float inv  = 1.0f / lrun[r];
        int   srow = qt * QB + wave * 16 + half * 8 + r;
#pragma unroll
        for (int f = 0; f < 4; ++f) {
            int d = h * HD_ + f * 16 + l16;
            o[((size_t)b * S_ + srow) * D_ + d] = (__bf16)(oacc[f][r] * inv);
        }
    }
}

// ---------------------------------------------------------------------------
extern "C" void kernel_launch(void* const* d_in, const int* in_sizes, int n_in,
                              void* d_out, int out_size, void* d_ws, size_t ws_size,
                              hipStream_t stream) {
    const float* x  = (const float*)d_in[0];
    const float* wq = (const float*)d_in[1];
    const float* bq = (const float*)d_in[2];
    const float* wk = (const float*)d_in[3];
    const float* bk = (const float*)d_in[4];
    const float* wv = (const float*)d_in[5];
    const float* bv = (const float*)d_in[6];
    const float* wo = (const float*)d_in[7];
    const float* bo = (const float*)d_in[8];

    const size_t MB = 1u << 20;
    char* ws = (char*)d_ws;
    __bf16* xb  = (__bf16*)(ws + 0 * MB);    // 16 MB
    __bf16* wqb = (__bf16*)(ws + 16 * MB);   //  2 MB
    __bf16* wkb = (__bf16*)(ws + 18 * MB);
    __bf16* wvb = (__bf16*)(ws + 20 * MB);
    __bf16* wob = (__bf16*)(ws + 22 * MB);
    __bf16* qb  = (__bf16*)(ws + 24 * MB);   // 16 MB each, [B,H,S,HD]
    __bf16* kb  = (__bf16*)(ws + 40 * MB);
    __bf16* vb  = (__bf16*)(ws + 56 * MB);
    __bf16* ob  = (__bf16*)(ws + 72 * MB);   // 16 MB, [B,S,D]

    const int nX = M_ * D_;        // 8M
    const int nW = D_ * D_;        // 1M

    attn_cast_bf16<<<nX / (256 * 8), 256, 0, stream>>>(x,  xb,  nX);
    attn_cast_bf16<<<nW / (256 * 8), 256, 0, stream>>>(wq, wqb, nW);
    attn_cast_bf16<<<nW / (256 * 8), 256, 0, stream>>>(wk, wkb, nW);
    attn_cast_bf16<<<nW / (256 * 8), 256, 0, stream>>>(wv, wvb, nW);
    attn_cast_bf16<<<nW / (256 * 8), 256, 0, stream>>>(wo, wob, nW);

    dim3 gg(M_ / 128, D_ / 128);   // 64 x 8
    attn_gemm_wmma<true,  false><<<gg, 256, 0, stream>>>(xb, wqb, bq, qb, M_, D_, D_);
    attn_gemm_wmma<true,  false><<<gg, 256, 0, stream>>>(xb, wkb, bk, kb, M_, D_, D_);
    attn_gemm_wmma<true,  false><<<gg, 256, 0, stream>>>(xb, wvb, bv, vb, M_, D_, D_);

    attn_flash_wmma<<<B_ * H_ * (S_ / 128), 256, 0, stream>>>(qb, kb, vb, ob);

    attn_gemm_wmma<false, true><<<gg, 256, 0, stream>>>(ob, wob, bo, d_out, M_, D_, D_);
}
